// KDAPolicyNetwork_43757126811711
// MI455X (gfx1250) — compile-verified, matrix operand order checked by reading, not run on hardware
//
#include <hip/hip_runtime.h>

#define T_LEN 32768
#define CH    64
#define NCH   512   // T_LEN / CH

typedef __attribute__((ext_vector_type(16))) _Float16 f16x16;
typedef __attribute__((ext_vector_type(8)))  float    f32x8;
typedef __attribute__((ext_vector_type(2)))  float    f32x2;
typedef unsigned int u32;
typedef __attribute__((ext_vector_type(4))) u32 u32x4;
typedef __attribute__((ext_vector_type(4))) int i32x4;
typedef __attribute__((ext_vector_type(8))) int i32x8;

__device__ __forceinline__ float sigm(float x)  { return 1.f / (1.f + __expf(-x)); }
__device__ __forceinline__ float siluf(float x) { return x * sigm(x); }
__device__ __forceinline__ float softpl(float x){ return x > 20.f ? x : log1pf(__expf(x)); }

// ---------------------------------------------------------------------------
// Tensor Data Mover: 1-D bulk copy of n contiguous f32 elements global -> LDS.
// D# built per CDNA5 ISA:
//   group0: count=1 | lds_addr | global_addr[56:0] | type=2
//   group1: data_size=4B, tensor_dim0=n, tensor_dim1=1, tile_dim0=n, tile_dim1=1
// Issued once per wave (EXEC ignored for TDM ops); completion via TENSORcnt.
// ---------------------------------------------------------------------------
__device__ __forceinline__ void tdm_load_f32(void* lds, const void* g, u32 n) {
#if defined(__HIP_DEVICE_COMPILE__)
  unsigned long long ga = (unsigned long long)(uintptr_t)g;
  u32x4 g0;
  g0[0] = 1u;                                  // count=1, user descriptor
  g0[1] = (u32)(uintptr_t)lds;                 // lds byte address
  g0[2] = (u32)ga;                             // global_addr[31:0]
  g0[3] = (u32)((ga >> 32) & 0x1ffffffu) | (2u << 30);  // addr[56:32] | type=2
  i32x8 g1;
  g1[0] = (int)(2u << 16);                     // data_size = 4 bytes
  g1[1] = (int)((n & 0xffffu) << 16);          // tensor_dim0[15:0]
  g1[2] = (int)(((n >> 16) & 0xffffu) | (1u << 16)); // tensor_dim0[31:16] | tensor_dim1=1
  g1[3] = (int)((n & 0xffffu) << 16);          // tile_dim0 = n
  g1[4] = 1;                                   // tile_dim1 = 1
  g1[5] = (int)n;                              // tensor_dim0_stride
  g1[6] = 0;
  g1[7] = 0;
  i32x4 z4 = {0, 0, 0, 0};
#if __clang_major__ >= 23
  i32x8 z8 = {0, 0, 0, 0, 0, 0, 0, 0};
  __builtin_amdgcn_tensor_load_to_lds(g0, g1, z4, z4, z8, 0);
#else
  __builtin_amdgcn_tensor_load_to_lds(g0, g1, z4, z4, 0);
#endif
#else
  (void)lds; (void)g; (void)n;
#endif
}
__device__ __forceinline__ void tdm_wait() {
#if defined(__HIP_DEVICE_COMPILE__)
  __builtin_amdgcn_s_wait_tensorcnt(0);
#endif
}

// ---------------------------------------------------------------------------
// Generic tiled WMMA GEMM:  C(T x N) = A(T x K) @ W(K x N), K%32==0, N%16==0.
// One wave32 per 64x16 output strip: the f16 weight fragment (B) is loaded
// once per k-step and reused across 4 v_wmma issues (4 M-tiles).
// ---------------------------------------------------------------------------
__global__ void k_gemm(const float* __restrict__ A, const _Float16* __restrict__ W,
                       float* __restrict__ C, int K, int N, int accumulate) {
  int ntiles = N >> 4;
  int tile = blockIdx.x;
  int tg = tile / ntiles;           // 64-token group
  int tn = tile % ntiles;
  int lane = threadIdx.x & 31;
  int m = lane & 15, hf = lane >> 4;
  const _Float16* Wc = W + tn * 16 + m;   // column (tn*16+m)
  f32x8 acc[4];
#pragma unroll
  for (int u = 0; u < 4; ++u)
#pragma unroll
    for (int r = 0; r < 8; ++r) acc[u][r] = 0.f;
  for (int k0 = 0; k0 < K; k0 += 32) {
    int kb = k0 + hf * 8;
    f16x16 b;
#pragma unroll
    for (int i = 0; i < 8; ++i) b[i]     = Wc[(size_t)(kb + i) * N];
#pragma unroll
    for (int i = 0; i < 8; ++i) b[8 + i] = Wc[(size_t)(kb + 16 + i) * N];
#pragma unroll
    for (int u = 0; u < 4; ++u) {
      const float* Arow = A + (size_t)(tg * 64 + u * 16 + m) * K;
      f16x16 a;
#pragma unroll
      for (int i = 0; i < 8; ++i) a[i]     = (_Float16)Arow[kb + i];
#pragma unroll
      for (int i = 0; i < 8; ++i) a[8 + i] = (_Float16)Arow[kb + 16 + i];
      acc[u] = __builtin_amdgcn_wmma_f32_16x16x32_f16(false, a, false, b, (short)0, acc[u], false, false);
    }
  }
#pragma unroll
  for (int u = 0; u < 4; ++u) {
    float* Cp = C + (size_t)(tg * 64 + u * 16 + hf * 8) * N + tn * 16 + m;
#pragma unroll
    for (int r = 0; r < 8; ++r) {
      float v = acc[u][r];
      if (accumulate) v += Cp[(size_t)r * N];
      Cp[(size_t)r * N] = v;
    }
  }
}

// ---------------------------------------------------------------------------
// Weight padding / conversion helpers
// ---------------------------------------------------------------------------
__global__ void k_zeroh(_Float16* p, int n) {
  int i = blockIdx.x * blockDim.x + threadIdx.x;
  if (i < n) p[i] = (_Float16)0.f;
}
__global__ void k_pad(const float* __restrict__ src, _Float16* __restrict__ dst,
                      int sk, int sn, int dn, int coff) {
  int i = blockIdx.x * blockDim.x + threadIdx.x;
  if (i >= sk * sn) return;
  int k = i / sn, n = i % sn;
  dst[(size_t)k * dn + coff + n] = (_Float16)src[i];
}

// ---------------------------------------------------------------------------
// Input projection: v0 = rms(silu(x@inp_w + b)) * inp_norm ; streams zeroed
// ---------------------------------------------------------------------------
__global__ void k_input(const float* __restrict__ x, const float* __restrict__ w,
                        const float* __restrict__ b, const float* __restrict__ nrm,
                        float* __restrict__ streamB, float* __restrict__ acc0) {
  int t = blockIdx.x * blockDim.x + threadIdx.x;
  if (t >= T_LEN) return;
  float xi[14];
#pragma unroll
  for (int i = 0; i < 14; ++i) xi[i] = x[(size_t)t * 14 + i];
  float y[32]; float ss = 0.f;
  for (int d = 0; d < 32; ++d) {
    float a = b[d];
    for (int i = 0; i < 14; ++i) a += xi[i] * w[i * 32 + d];
    a = siluf(a); y[d] = a; ss += a * a;
  }
  float sc = rsqrtf(ss / 32.f + 1.1920929e-7f);
  for (int d = 0; d < 32; ++d) {
    float v = y[d] * sc * nrm[d];
    streamB[(size_t)t * 32 + d] = v;
    acc0[(size_t)t * 32 + d] = v;
    streamB[((size_t)1 * T_LEN + t) * 32 + d] = 0.f;
    streamB[((size_t)2 * T_LEN + t) * 32 + d] = 0.f;
    streamB[((size_t)3 * T_LEN + t) * 32 + d] = 0.f;
  }
}

// ---------------------------------------------------------------------------
// mHC: build normalized concat x (T x 128); one wave per token, lane = dim
// ---------------------------------------------------------------------------
__global__ void k_mhc_x(const float* __restrict__ streamB, float* __restrict__ xcat) {
  int lane = threadIdx.x & 31;
  int t = blockIdx.x * 8 + (threadIdx.x >> 5);
  float s[4]; float ss = 0.f;
#pragma unroll
  for (int n = 0; n < 4; ++n) { s[n] = streamB[((size_t)n * T_LEN + t) * 32 + lane]; ss += s[n] * s[n]; }
  for (int o = 16; o >= 1; o >>= 1) ss += __shfl_xor(ss, o, 32);
  float sc = rsqrtf(ss / 128.f + 1e-8f);
#pragma unroll
  for (int n = 0; n < 4; ++n) xcat[(size_t)t * 128 + n * 32 + lane] = s[n] * sc;
}

// mHC apply: sinkhorn(4x4, 20 iters) + stream mixing + H_pre combine.
__global__ void k_mhc_apply(float* __restrict__ streamB, const float* __restrict__ raw,
                            float* __restrict__ h, float* __restrict__ Hpost,
                            const float* a_pre, const float* a_post, const float* a_res,
                            const float* b_pre, const float* b_post, const float* b_res) {
  int lane = threadIdx.x & 31;
  int t = blockIdx.x * 8 + (threadIdx.x >> 5);
  const float* r = raw + (size_t)t * 32;
  float ap = a_pre[0], apo = a_post[0], ar = a_res[0];
  float Hpre[4], Hpo[4], M[16];
#pragma unroll
  for (int n = 0; n < 4; ++n) {
    Hpre[n] = sigm(ap * r[n] + b_pre[n]);
    Hpo[n]  = 2.f * sigm(apo * r[4 + n] + b_post[n]);
  }
#pragma unroll
  for (int i = 0; i < 16; ++i) M[i] = __expf(ar * r[8 + i] + b_res[i]);
  for (int it = 0; it < 20; ++it) {
#pragma unroll
    for (int i = 0; i < 4; ++i) {
      float s = M[i*4] + M[i*4+1] + M[i*4+2] + M[i*4+3];
      M[i*4] /= s; M[i*4+1] /= s; M[i*4+2] /= s; M[i*4+3] /= s;
    }
#pragma unroll
    for (int j = 0; j < 4; ++j) {
      float s = M[j] + M[4+j] + M[8+j] + M[12+j];
      M[j] /= s; M[4+j] /= s; M[8+j] /= s; M[12+j] /= s;
    }
  }
  float old[4], nw[4];
#pragma unroll
  for (int n = 0; n < 4; ++n) old[n] = streamB[((size_t)n * T_LEN + t) * 32 + lane];
  float hv = 0.f;
#pragma unroll
  for (int i = 0; i < 4; ++i) {
    float a = 0.f;
#pragma unroll
    for (int j = 0; j < 4; ++j) a += M[i*4 + j] * old[j];
    nw[i] = a; hv += Hpre[i] * a;
  }
#pragma unroll
  for (int n = 0; n < 4; ++n) streamB[((size_t)n * T_LEN + t) * 32 + lane] = nw[n];
  h[(size_t)t * 32 + lane] = hv;
  if (lane < 4) Hpost[(size_t)t * 4 + lane] = Hpo[lane];
}

// ---------------------------------------------------------------------------
// Depth attention over accumulated outputs (softmax over layers per token)
// ---------------------------------------------------------------------------
__global__ void k_attnres(float* __restrict__ h, const float* __restrict__ Kc,
                          const float* __restrict__ acc, const float* __restrict__ wl,
                          int nacc, int overwrite) {
  int t = blockIdx.x * blockDim.x + threadIdx.x;
  if (t >= T_LEN) return;
  float wv[32];
#pragma unroll
  for (int d = 0; d < 32; ++d) wv[d] = wl[d];
  float lg[17]; float mx = -1e30f;
  for (int l = 0; l < nacc; ++l) {
    const float* kp = Kc + ((size_t)l * T_LEN + t) * 32;
    float s = 0.f;
    for (int d = 0; d < 32; ++d) s += wv[d] * kp[d];
    lg[l] = s; mx = fmaxf(mx, s);
  }
  float den = 0.f;
  for (int l = 0; l < nacc; ++l) { lg[l] = __expf(lg[l] - mx); den += lg[l]; }
  float inv = 1.f / den;
  float sv[32];
#pragma unroll
  for (int d = 0; d < 32; ++d) sv[d] = 0.f;
  for (int l = 0; l < nacc; ++l) {
    float a = lg[l] * inv;
    const float* vp = acc + ((size_t)l * T_LEN + t) * 32;
    for (int d = 0; d < 32; ++d) sv[d] += a * vp[d];
  }
  for (int d = 0; d < 32; ++d) {
    if (overwrite) h[(size_t)t * 32 + d] = sv[d];
    else           h[(size_t)t * 32 + d] += sv[d];
  }
}

// Generic weighted RMS norm over 32 dims.
__global__ void k_rmsw(const float* __restrict__ in, const float* __restrict__ w,
                       float* __restrict__ out) {
  int t = blockIdx.x * blockDim.x + threadIdx.x;
  if (t >= T_LEN) return;
  float v[32]; float ss = 0.f;
  for (int d = 0; d < 32; ++d) { v[d] = in[(size_t)t * 32 + d]; ss += v[d] * v[d]; }
  float sc = rsqrtf(ss / 32.f + 1.1920929e-7f);
  for (int d = 0; d < 32; ++d) out[(size_t)t * 32 + d] = v[d] * sc * w[d];
}

// ---------------------------------------------------------------------------
// KDA elementwise prep: normalize+PoPE q/k, silu v, alpha-gate input g, Wd-gate
// ---------------------------------------------------------------------------
__global__ void k_kda_prep(const float* __restrict__ raw4, const float* __restrict__ agraw,
                           const float* __restrict__ delta,
                           float* __restrict__ qb, float* __restrict__ kb,
                           float* __restrict__ vb, float* __restrict__ gb,
                           float* __restrict__ dgb) {
  int t = blockIdx.x * blockDim.x + threadIdx.x;
  if (t >= T_LEN) return;
  const float* r = raw4 + (size_t)t * 64;
  float q[16], k[16], nq = 0.f, nk = 0.f;
#pragma unroll
  for (int i = 0; i < 16; ++i) { q[i] = r[i]; k[i] = r[16 + i]; nq += q[i]*q[i]; nk += k[i]*k[i]; }
  nq = 1.f / fmaxf(sqrtf(nq), 1e-12f);
  nk = 1.f / fmaxf(sqrtf(nk), 1e-12f);
  float pos = (float)t;
#pragma unroll
  for (int i = 0; i < 16; ++i) {
    float mq = softpl(q[i] * nq), mk = softpl(k[i] * nk);
    float fr = __powf(10000.f, (float)i / 16.f);
    float ph = pos * fr;
    float phk = ph - 6.2831853071795864f * sigm(delta[i]);
    qb[(size_t)t * 32 + i]      = mq * __cosf(ph);
    qb[(size_t)t * 32 + 16 + i] = mq * __sinf(ph);
    kb[(size_t)t * 32 + i]      = mk * __cosf(phk);
    kb[(size_t)t * 32 + 16 + i] = mk * __sinf(phk);
  }
#pragma unroll
  for (int i = 0; i < 16; ++i) vb[(size_t)t * 16 + i] = siluf(r[32 + i]);
  const float* g = agraw + (size_t)t * 64;
#pragma unroll
  for (int i = 0; i < 32; ++i) gb[(size_t)t * 32 + i] = siluf(g[i]) * g[32 + i];
#pragma unroll
  for (int i = 0; i < 16; ++i) dgb[(size_t)t * 32 + i] = siluf(r[48 + i]);
#pragma unroll
  for (int i = 16; i < 32; ++i) dgb[(size_t)t * 32 + i] = 0.f;
}

__global__ void k_sigK(const float* __restrict__ in, float* __restrict__ out, int n) {
  int i = blockIdx.x * blockDim.x + threadIdx.x;
  if (i < n) out[i] = sigm(in[i]);
}

// ---------------------------------------------------------------------------
// KDA phase A: per-chunk transition M (32x32), N (32x16). One wave per chunk.
// Whole chunk's k/a/v staged into LDS by the Tensor Data Mover up front:
// the 64-step recurrence then runs with zero global loads / zero barriers.
// ---------------------------------------------------------------------------
__global__ void k_chunkMN(const float* __restrict__ kb, const float* __restrict__ ab,
                          const float* __restrict__ vb,
                          float* __restrict__ Mb, float* __restrict__ Nb) {
  __shared__ float sK[CH * 32], sA[CH * 32], sW[CH * 32], sV[CH * 16];
  int lane = threadIdx.x;
  int c = blockIdx.x;
  tdm_load_f32(sK, kb + (size_t)c * CH * 32, CH * 32);
  tdm_load_f32(sA, ab + (size_t)c * CH * 32, CH * 32);
  tdm_load_f32(sV, vb + (size_t)c * CH * 16, CH * 16);
  tdm_wait();
  __syncthreads();
  for (int s = 0; s < CH; ++s) sW[s * 32 + lane] = sK[s * 32 + lane] * sA[s * 32 + lane];
  __syncthreads();
  int jn = lane & 15;
  float Mcol[32], Ncol[32];
#pragma unroll
  for (int i = 0; i < 32; ++i) { Mcol[i] = (i == lane) ? 1.f : 0.f; Ncol[i] = 0.f; }
  for (int s = 0; s < CH; ++s) {
    const float* kp = sK + s * 32;
    const float* apt = sA + s * 32;
    const float* wp = sW + s * 32;
    float r = 0.f, rn = 0.f;
#pragma unroll
    for (int i = 0; i < 32; ++i) { r += wp[i] * Mcol[i]; rn += wp[i] * Ncol[i]; }
    float vj = sV[s * 16 + jn];
#pragma unroll
    for (int i = 0; i < 32; ++i) {
      Mcol[i] = apt[i] * Mcol[i] - kp[i] * r;
      Ncol[i] = apt[i] * Ncol[i] + kp[i] * (vj - rn);
    }
  }
#pragma unroll
  for (int i = 0; i < 32; ++i) Mb[(size_t)c * 1024 + i * 32 + lane] = Mcol[i];
  if (lane < 16)
#pragma unroll
    for (int i = 0; i < 32; ++i) Nb[(size_t)c * 512 + i * 16 + lane] = Ncol[i];
}

// ---------------------------------------------------------------------------
// KDA phase B: serial chunk scan  S <- M_c @ S + N_c  via v_wmma_f32_16x16x4_f32.
// Single wave; next chunk's M/N prefetched (global_prefetch) during compute.
// ---------------------------------------------------------------------------
__global__ void k_scan(const float* __restrict__ Mb, const float* __restrict__ Nb,
                       float* __restrict__ Sc) {
  __shared__ float sS[512];
  int lane = threadIdx.x;
  int m = lane & 15, hf = lane >> 4;
  f32x8 s0 = {0.f,0.f,0.f,0.f,0.f,0.f,0.f,0.f};
  f32x8 s1 = {0.f,0.f,0.f,0.f,0.f,0.f,0.f,0.f};
  for (int c = 0; c < NCH; ++c) {
    if (c + 1 < NCH) {
      __builtin_prefetch((const void*)(Mb + (size_t)(c + 1) * 1024), 0, 1);
      __builtin_prefetch((const void*)(Nb + (size_t)(c + 1) * 512), 0, 1);
    }
    float* sp = Sc + (size_t)c * 512;
#pragma unroll
    for (int r2 = 0; r2 < 8; ++r2) {
      int row = hf * 8 + r2;
      sp[row * 16 + m] = s0[r2];        sp[(16 + row) * 16 + m] = s1[r2];
      sS[row * 16 + m] = s0[r2];        sS[(16 + row) * 16 + m] = s1[r2];
    }
    __syncthreads();
    const float* Mc = Mb + (size_t)c * 1024;
    const float* Nc = Nb + (size_t)c * 512;
    f32x8 c0, c1;
#pragma unroll
    for (int r2 = 0; r2 < 8; ++r2) {
      int row = hf * 8 + r2;
      c0[r2] = Nc[row * 16 + m];
      c1[r2] = Nc[(16 + row) * 16 + m];
    }
#pragma unroll
    for (int k0 = 0; k0 < 32; k0 += 4) {
      int ks = k0 + hf * 2;
      f32x2 b;  b[0]  = sS[ks * 16 + m];        b[1]  = sS[(ks + 1) * 16 + m];
      f32x2 a0; a0[0] = Mc[m * 32 + ks];        a0[1] = Mc[m * 32 + ks + 1];
      f32x2 a1; a1[0] = Mc[(16 + m) * 32 + ks]; a1[1] = Mc[(16 + m) * 32 + ks + 1];
      c0 = __builtin_amdgcn_wmma_f32_16x16x4_f32(false, a0, false, b, (short)0, c0, false, false);
      c1 = __builtin_amdgcn_wmma_f32_16x16x4_f32(false, a1, false, b, (short)0, c1, false, false);
    }
    s0 = c0; s1 = c1;
    __syncthreads();
  }
}

// ---------------------------------------------------------------------------
// KDA phase C: replay within chunk from stored chunk-start S; out_t = q_t @ S_t
// Chunk inputs (q,k,a,v) TDM-staged into LDS; barrier-free inner loop.
// ---------------------------------------------------------------------------
__global__ void k_replay(const float* __restrict__ qb, const float* __restrict__ kb,
                         const float* __restrict__ ab, const float* __restrict__ vb,
                         const float* __restrict__ Sc, float* __restrict__ outRaw) {
  __shared__ float sK[CH * 32], sA[CH * 32], sW[CH * 32], sQ[CH * 32], sV[CH * 16];
  int lane = threadIdx.x;
  int c = blockIdx.x;
  tdm_load_f32(sK, kb + (size_t)c * CH * 32, CH * 32);
  tdm_load_f32(sA, ab + (size_t)c * CH * 32, CH * 32);
  tdm_load_f32(sQ, qb + (size_t)c * CH * 32, CH * 32);
  tdm_load_f32(sV, vb + (size_t)c * CH * 16, CH * 16);
  tdm_wait();
  __syncthreads();
  for (int s = 0; s < CH; ++s) sW[s * 32 + lane] = sK[s * 32 + lane] * sA[s * 32 + lane];
  __syncthreads();
  int j = lane & 15, hf = lane >> 4;
  float s[16];
  const float* sp = Sc + (size_t)c * 512;
#pragma unroll
  for (int i = 0; i < 16; ++i) s[i] = sp[(hf * 16 + i) * 16 + j];
  for (int st = 0; st < CH; ++st) {
    const float* kp = sK + st * 32;
    const float* apt = sA + st * 32;
    const float* wp = sW + st * 32;
    const float* qp = sQ + st * 32;
    float rn = 0.f;
#pragma unroll
    for (int i = 0; i < 16; ++i) rn += wp[hf * 16 + i] * s[i];
    rn += __shfl_xor(rn, 16, 32);
    float vj = sV[st * 16 + j];
    float op = 0.f;
#pragma unroll
    for (int i = 0; i < 16; ++i) {
      int rr = hf * 16 + i;
      s[i] = apt[rr] * s[i] + kp[rr] * (vj - rn);
      op += qp[rr] * s[i];
    }
    op += __shfl_xor(op, 16, 32);
    if (lane < 16) outRaw[((size_t)c * CH + st) * 16 + j] = op;
  }
}

// KDA post: rms(out)*pn * sigmoid(u); zero-padded to K=32 for Wout WMMA.
__global__ void k_kda_post(const float* __restrict__ outRaw, const float* __restrict__ uRaw,
                           const float* __restrict__ pn, float* __restrict__ outPad) {
  int t = blockIdx.x * blockDim.x + threadIdx.x;
  if (t >= T_LEN) return;
  float o[16]; float ss = 0.f;
#pragma unroll
  for (int i = 0; i < 16; ++i) { o[i] = outRaw[(size_t)t * 16 + i]; ss += o[i] * o[i]; }
  float sc = rsqrtf(ss / 16.f + 1.1920929e-7f);
#pragma unroll
  for (int i = 0; i < 16; ++i)
    outPad[(size_t)t * 32 + i] = o[i] * sc * pn[i] * sigm(uRaw[(size_t)t * 16 + i]);
#pragma unroll
  for (int i = 16; i < 32; ++i) outPad[(size_t)t * 32 + i] = 0.f;
}

// stream[n,t,:] += Hpost[t,n] * out[t,:]
__global__ void k_stream_post(float* __restrict__ streamB, const float* __restrict__ Hpost,
                              const float* __restrict__ outD) {
  int idx = blockIdx.x * blockDim.x + threadIdx.x;
  if (idx >= T_LEN * 32) return;
  int t = idx >> 5;
  float o = outD[idx];
#pragma unroll
  for (int n = 0; n < 4; ++n)
    streamB[(size_t)n * T_LEN * 32 + idx] += Hpost[(size_t)t * 4 + n] * o;
}

// SwiGLU elementwise pieces
__global__ void k_swi_e1(const float* __restrict__ raw1, float* __restrict__ gd) {
  int i = blockIdx.x * blockDim.x + threadIdx.x;
  if (i >= T_LEN * 32) return;
  int t = i >> 5, d = i & 31;
  gd[i] = siluf(raw1[(size_t)t * 160 + d]);
}
__global__ void k_swi_e2(const float* __restrict__ raw1, const float* __restrict__ wuRaw,
                         float* __restrict__ act) {
  int i = blockIdx.x * blockDim.x + threadIdx.x;
  if (i >= T_LEN * 64) return;
  int t = i >> 6, j = i & 63;
  act[i] = sigm(wuRaw[i]) * siluf(raw1[(size_t)t * 160 + 32 + j]) * raw1[(size_t)t * 160 + 96 + j];
}

// Head elementwise + final writeout (11 logits per token)
__global__ void k_head_e(const float* __restrict__ hr, float* __restrict__ hh) {
  int i = blockIdx.x * blockDim.x + threadIdx.x;
  if (i >= T_LEN * 64) return;
  int t = i >> 6, j = i & 63;
  hh[i] = siluf(hr[(size_t)t * 128 + j]) * hr[(size_t)t * 128 + 64 + j];
}
__global__ void k_writeout(const float* __restrict__ src, float* __restrict__ dst) {
  int i = blockIdx.x * blockDim.x + threadIdx.x;
  if (i >= T_LEN * 11) return;
  int t = i / 11, j = i % 11;
  dst[i] = src[(size_t)t * 16 + j];
}

// ===========================================================================
extern "C" void kernel_launch(void* const* d_in, const int* in_sizes, int n_in,
                              void* d_out, int out_size, void* d_ws, size_t ws_size,
                              hipStream_t stream) {
  (void)in_sizes; (void)n_in; (void)out_size; (void)ws_size;
  // ---- input mapping: x first, then params leaves (dict keys sorted, lists in order)
  int ii = 0;
  auto F = [&]() { return (const float*)d_in[ii++]; };
  const float* X         = F();      // (T,14)
  const float* attn_norm = F();      // (32)
  const float* head_down = F();      // (51,11)
  const float* head_gate = F();      // (32,51)
  const float* head_up   = F();      // (32,51)
  const float* inp_b     = F();      // (32)
  const float* inp_norm  = F();      // (32)
  const float* inp_w     = F();      // (14,32)
  struct KdaP { const float *Wd,*Wk,*Wout,*Wq,*Wu,*Wv,*ad,*ag,*au,*delta,*pn; } KD[8];
  for (int l = 0; l < 8; ++l) KD[l] = KdaP{F(),F(),F(),F(),F(),F(),F(),F(),F(),F(),F()};
  struct MhcP { const float *a_post,*a_pre,*a_res,*b_post,*b_pre,*b_res,*phi_post,*phi_pre,*phi_res; } MH[16];
  for (int m = 0; m < 16; ++m) MH[m] = MhcP{F(),F(),F(),F(),F(),F(),F(),F(),F()};
  const float* pre_w = F();          // (32,32)
  struct SwiP { const float *down,*gate,*norm,*up,*wd,*wu; } SW[8];
  for (int l = 0; l < 8; ++l) SW[l] = SwiP{F(),F(),F(),F(),F(),F()};
  const float* Wl[17];
  for (int i = 0; i < 17; ++i) Wl[i] = F();

  // ---- workspace bump allocator
  char* base = (char*)d_ws; size_t off = 0;
  auto AF = [&](size_t n) { float* p = (float*)(base + off); off += n * 4; return p; };
  float* streamB = AF((size_t)4 * T_LEN * 32);
  float* accB    = AF((size_t)17 * T_LEN * 32);
  float* KcB     = AF((size_t)17 * T_LEN * 32);
  float* xcat    = AF((size_t)T_LEN * 128);
  float* hB      = AF((size_t)T_LEN * 32);
  float* hn      = AF((size_t)T_LEN * 32);
  float* mhcRaw  = AF((size_t)T_LEN * 32);
  float* HpostB  = AF((size_t)T_LEN * 4);
  float* kdaRaw  = AF((size_t)T_LEN * 64);
  float* agRaw   = AF((size_t)T_LEN * 64);
  float* qb      = AF((size_t)T_LEN * 32);
  float* kb      = AF((size_t)T_LEN * 32);
  float* vb      = AF((size_t)T_LEN * 16);
  float* gb      = AF((size_t)T_LEN * 32);
  float* dgb     = AF((size_t)T_LEN * 32);
  float* alrB    = AF((size_t)T_LEN * 32);
  float* alB     = AF((size_t)T_LEN * 32);
  float* uRaw    = AF((size_t)T_LEN * 16);
  float* Mb      = AF((size_t)NCH * 1024);
  float* Nb      = AF((size_t)NCH * 512);
  float* ScB     = AF((size_t)NCH * 512);
  float* outRaw  = AF((size_t)T_LEN * 16);
  float* outPad  = AF((size_t)T_LEN * 32);
  float* rmstmp  = AF((size_t)T_LEN * 32);
  float* swiRaw  = AF((size_t)T_LEN * 160);
  float* gdB     = AF((size_t)T_LEN * 32);
  float* wuRawB  = AF((size_t)T_LEN * 64);
  float* actB    = AF((size_t)T_LEN * 64);
  float* headOut = AF((size_t)T_LEN * 16);
  float* headRaw = xcat;   // reuse: head phase only
  float* hhB     = agRaw;  // reuse: head phase only
  auto AH = [&](size_t n) { _Float16* p = (_Float16*)(base + off); off += n * 2; return p; };
  _Float16* h16base = (_Float16*)(base + off);
  _Float16* WqkvD = AH((size_t)8 * 32 * 64);
  _Float16* AgAu  = AH((size_t)8 * 32 * 64);
  _Float16* AdP   = AH((size_t)8 * 32 * 32);
  _Float16* WuKP  = AH((size_t)8 * 32 * 16);
  _Float16* WoutP = AH((size_t)8 * 32 * 32);
  _Float16* PhiC  = AH((size_t)16 * 128 * 32);
  _Float16* W1S   = AH((size_t)8 * 32 * 160);
  _Float16* WuS   = AH((size_t)8 * 32 * 64);
  _Float16* DownP = AH((size_t)8 * 64 * 32);
  _Float16* HGU   = AH((size_t)32 * 128);
  _Float16* HDp   = AH((size_t)64 * 16);
  _Float16* PreWh = AH((size_t)32 * 32);
  int nHalves = (int)((_Float16*)(base + off) - h16base);

  // ---- weight prep (every call; deterministic)
  k_zeroh<<<(nHalves + 255) / 256, 256, 0, stream>>>(h16base, nHalves);
  auto pad = [&](const float* src, int sk, int sn, _Float16* dst, int dn, int coff) {
    int n = sk * sn;
    k_pad<<<(n + 255) / 256, 256, 0, stream>>>(src, dst, sk, sn, dn, coff);
  };
  for (int l = 0; l < 8; ++l) {
    pad(KD[l].Wq,   32, 16, WqkvD + l * 2048, 64, 0);
    pad(KD[l].Wk,   32, 16, WqkvD + l * 2048, 64, 16);
    pad(KD[l].Wv,   32, 16, WqkvD + l * 2048, 64, 32);
    pad(KD[l].Wd,   32,  9, WqkvD + l * 2048, 64, 48);
    pad(KD[l].ag,   32, 25, AgAu  + l * 2048, 64, 0);
    pad(KD[l].au,   32, 25, AgAu  + l * 2048, 64, 32);
    pad(KD[l].ad,   25, 32, AdP   + l * 1024, 32, 0);
    pad(KD[l].Wu,    9, 16, WuKP  + l * 512,  16, 0);
    pad(KD[l].Wout, 16, 32, WoutP + l * 1024, 32, 0);
    pad(SW[l].wd,   32, 32, W1S   + l * 5120, 160, 0);
    pad(SW[l].gate, 32, 51, W1S   + l * 5120, 160, 32);
    pad(SW[l].up,   32, 51, W1S   + l * 5120, 160, 96);
    pad(SW[l].wu,   32, 51, WuS   + l * 2048, 64, 0);
    pad(SW[l].down, 51, 32, DownP + l * 2048, 32, 0);
  }
  for (int m = 0; m < 16; ++m) {
    pad(MH[m].phi_pre,  128,  4, PhiC + (size_t)m * 4096, 32, 0);
    pad(MH[m].phi_post, 128,  4, PhiC + (size_t)m * 4096, 32, 4);
    pad(MH[m].phi_res,  128, 16, PhiC + (size_t)m * 4096, 32, 8);
  }
  pad(head_gate, 32, 51, HGU, 128, 0);
  pad(head_up,   32, 51, HGU, 128, 64);
  pad(head_down, 51, 11, HDp, 16, 0);
  pad(pre_w,     32, 32, PreWh, 32, 0);

  auto gemm = [&](const float* A, const _Float16* W, float* C, int K, int N, int acc) {
    k_gemm<<<(T_LEN / 64) * (N / 16), 32, 0, stream>>>(A, W, C, K, N, acc);
  };
  auto runMhc = [&](int m) {
    k_mhc_x<<<T_LEN / 8, 256, 0, stream>>>(streamB, xcat);
    gemm(xcat, PhiC + (size_t)m * 4096, mhcRaw, 128, 32, 0);
    k_mhc_apply<<<T_LEN / 8, 256, 0, stream>>>(streamB, mhcRaw, hB, HpostB,
        MH[m].a_pre, MH[m].a_post, MH[m].a_res, MH[m].b_pre, MH[m].b_post, MH[m].b_res);
  };
  auto attnres = [&](const float* wl, int nacc, int ow) {
    k_attnres<<<T_LEN / 256, 256, 0, stream>>>(hB, KcB, accB, wl, nacc, ow);
  };
  auto kcache = [&](const float* V, int idx) {
    k_rmsw<<<T_LEN / 256, 256, 0, stream>>>(V, attn_norm, rmstmp);
    gemm(rmstmp, PreWh, KcB + (size_t)idx * T_LEN * 32, 32, 32, 0);
  };

  // ---- forward ----
  k_input<<<T_LEN / 256, 256, 0, stream>>>(X, inp_w, inp_b, inp_norm, streamB, accB);
  kcache(accB, 0);
  int accN = 1;
  for (int l = 0; l < 8; ++l) {
    // ---------------- KDA sublayer ----------------
    runMhc(2 * l);
    if (accN > 1) attnres(Wl[2 * l], accN, 0);
    gemm(hB, WqkvD + l * 2048, kdaRaw, 32, 64, 0);
    gemm(hB, AgAu  + l * 2048, agRaw,  32, 64, 0);
    k_kda_prep<<<T_LEN / 256, 256, 0, stream>>>(kdaRaw, agRaw, KD[l].delta, qb, kb, vb, gb, dgb);
    gemm(gb,  AdP  + l * 1024, alrB, 32, 32, 0);
    k_sigK<<<(T_LEN * 32) / 256, 256, 0, stream>>>(alrB, alB, T_LEN * 32);
    gemm(dgb, WuKP + l * 512,  uRaw, 32, 16, 0);
    k_chunkMN<<<NCH, 32, 0, stream>>>(kb, alB, vb, Mb, Nb);
    k_scan<<<1, 32, 0, stream>>>(Mb, Nb, ScB);
    k_replay<<<NCH, 32, 0, stream>>>(qb, kb, alB, vb, ScB, outRaw);
    k_kda_post<<<T_LEN / 256, 256, 0, stream>>>(outRaw, uRaw, KD[l].pn, outPad);
    float* slot = accB + (size_t)accN * T_LEN * 32;
    gemm(outPad, WoutP + l * 1024, slot, 32, 32, 0);
    k_stream_post<<<(T_LEN * 32) / 256, 256, 0, stream>>>(streamB, HpostB, slot);
    kcache(slot, accN);
    accN++;
    // ---------------- SwiGLU sublayer ----------------
    runMhc(2 * l + 1);
    attnres(Wl[2 * l + 1], accN, 0);
    k_rmsw<<<T_LEN / 256, 256, 0, stream>>>(hB, SW[l].norm, hn);
    gemm(hn, W1S + l * 5120, swiRaw, 32, 160, 0);
    k_swi_e1<<<(T_LEN * 32) / 256, 256, 0, stream>>>(swiRaw, gdB);
    gemm(gdB, WuS + l * 2048, wuRawB, 32, 64, 0);
    k_swi_e2<<<(T_LEN * 64) / 256, 256, 0, stream>>>(swiRaw, wuRawB, actB);
    slot = accB + (size_t)accN * T_LEN * 32;
    gemm(actB, DownP + l * 2048, slot, 64, 32, 0);
    k_stream_post<<<(T_LEN * 32) / 256, 256, 0, stream>>>(streamB, HpostB, slot);
    kcache(slot, accN);
    accN++;
  }
  // ---------------- head ----------------
  attnres(Wl[16], accN, 1);
  gemm(hB, HGU, headRaw, 32, 128, 0);
  k_head_e<<<(T_LEN * 64) / 256, 256, 0, stream>>>(headRaw, hhB);
  gemm(hhB, HDp, headOut, 64, 16, 0);
  k_writeout<<<(T_LEN * 11 + 255) / 256, 256, 0, stream>>>(headOut, (float*)d_out);
}
